// GAT_76244259439101
// MI455X (gfx1250) — matrix operand order repaired and checked by reading, compiled, not run on hardware
//
#include <hip/hip_runtime.h>

#define N_NODES 50000
#define E_EDGES 800000
#define F_IN    602
#define K1P     608     // 19 * 32, padded K for layer-1 GEMM
#define KT1     19
#define C1      256     // HEADS * HID
#define NT1     16
#define HEADS   4
#define OUT_C   41
#define N2P     48      // 3 * 16, padded out cols for layer-2 GEMM
#define NT2     3
#define K2P     256
#define KT2     8
#define MT      3125    // 50000 / 16 M tiles
#define NEG_SLOPE 0.2f

typedef __attribute__((ext_vector_type(16))) _Float16 v16h;
typedef __attribute__((ext_vector_type(8)))  float    v8f;

// ---------------------------------------------------------------------------
// A-operand swizzle: fp32 [rows x cols] -> f16 fragments [mt][kt][lane][16]
// ISA A layout (16x32 f16): lane L (row = L&15), halves j<8 -> K = kbase+j,
// j>=8 -> K = 16+kbase+(j-8), kbase = (L<16)?0:8.
// ---------------------------------------------------------------------------
__global__ void k_swz_a(const float* __restrict__ src, _Float16* __restrict__ dst,
                        int rows, int cols, int ktiles, long long total) {
  long long i = (long long)blockIdx.x * blockDim.x + threadIdx.x;
  if (i >= total) return;
  int j    = (int)(i & 15);
  int lane = (int)((i >> 4) & 31);
  int kt   = (int)((i >> 9) % ktiles);
  long long mt = i / ((long long)ktiles << 9);
  int r = (int)(mt * 16 + (lane & 15));
  int kbase = (lane < 16) ? 0 : 8;
  int k = kt * 32 + ((j < 8) ? (kbase + j) : (16 + kbase + (j - 8)));
  float v = (r < rows && k < cols) ? src[(long long)r * cols + k] : 0.0f;
  dst[i] = (_Float16)v;
}

// ---------------------------------------------------------------------------
// B-operand swizzle: fp32 [rows x cols] -> f16 fragments [kt][nt][lane][16]
// ISA B layout (32x16 f16): lane L (col n = L&15), halves j<8 -> K = kbase+j,
// j>=8 -> K = 16+kbase+(j-8), kbase = (L<16)?0:8.
// ---------------------------------------------------------------------------
__global__ void k_swz_b(const float* __restrict__ src, _Float16* __restrict__ dst,
                        int rows, int cols, int ntiles, long long total) {
  long long i = (long long)blockIdx.x * blockDim.x + threadIdx.x;
  if (i >= total) return;
  int j    = (int)(i & 15);
  int lane = (int)((i >> 4) & 31);
  int nt   = (int)((i >> 9) % ntiles);
  int kt   = (int)(i / ((long long)ntiles << 9));
  int k = kt * 32 + ((j < 8) ? (((lane < 16) ? 0 : 8) + j) : (16 + ((lane < 16) ? 0 : 8) + (j - 8)));
  int n = nt * 16 + (lane & 15);
  float v = (k < rows && n < cols) ? src[(long long)k * cols + n] : 0.0f;
  dst[i] = (_Float16)v;
}

// ---------------------------------------------------------------------------
// Fused ELU + A-operand swizzle for the layer-2 input (rows=N, cols=256).
// ---------------------------------------------------------------------------
__global__ void k_elu_swz(const float* __restrict__ src, _Float16* __restrict__ dst,
                          int rows, int cols, int ktiles, long long total) {
  long long i = (long long)blockIdx.x * blockDim.x + threadIdx.x;
  if (i >= total) return;
  int j    = (int)(i & 15);
  int lane = (int)((i >> 4) & 31);
  int kt   = (int)((i >> 9) % ktiles);
  long long mt = i / ((long long)ktiles << 9);
  int r = (int)(mt * 16 + (lane & 15));
  int kbase = (lane < 16) ? 0 : 8;
  int k = kt * 32 + ((j < 8) ? (kbase + j) : (16 + kbase + (j - 8)));
  float v = (r < rows && k < cols) ? src[(long long)r * cols + k] : 0.0f;
  v = (v > 0.0f) ? v : (expf(v) - 1.0f);
  dst[i] = (_Float16)v;
}

// ---------------------------------------------------------------------------
// WMMA GEMM on pre-swizzled fragments.
// Each wave: 4 consecutive M tiles x 1 N tile, one B fragment reused 4x.
// block = 256 (8 waves, 8 N tiles); grid = (ceil(nt/8), ceil(mt/4)).
// ---------------------------------------------------------------------------
__global__ void k_gemm_wmma_f16(const _Float16* __restrict__ Asw,
                                const _Float16* __restrict__ Bsw,
                                float* __restrict__ C,
                                int mtiles, int ntiles, int ktiles, int Np) {
  const int lane = threadIdx.x & 31;
  const int tn = blockIdx.x * (blockDim.x >> 5) + (threadIdx.x >> 5);
  if (tn >= ntiles) return;
  const int mt0 = blockIdx.y * 4;

  const long long mstep = (long long)ktiles << 9;   // halves per M tile
  const long long bstep = (long long)ntiles << 9;   // halves per K tile of B
  const _Float16* __restrict__ ap0 =
      Asw + (long long)((mt0 + 0 < mtiles) ? mt0 + 0 : mtiles - 1) * mstep + lane * 16;
  const _Float16* __restrict__ ap1 =
      Asw + (long long)((mt0 + 1 < mtiles) ? mt0 + 1 : mtiles - 1) * mstep + lane * 16;
  const _Float16* __restrict__ ap2 =
      Asw + (long long)((mt0 + 2 < mtiles) ? mt0 + 2 : mtiles - 1) * mstep + lane * 16;
  const _Float16* __restrict__ ap3 =
      Asw + (long long)((mt0 + 3 < mtiles) ? mt0 + 3 : mtiles - 1) * mstep + lane * 16;
  const _Float16* __restrict__ bp = Bsw + (long long)tn * 512 + lane * 16;

  v8f acc0 = {}, acc1 = {}, acc2 = {}, acc3 = {};
  for (int kt = 0; kt < ktiles; ++kt) {
    v16h b  = *(const v16h*)bp;
    v16h a0 = *(const v16h*)ap0;
    v16h a1 = *(const v16h*)ap1;
    v16h a2 = *(const v16h*)ap2;
    v16h a3 = *(const v16h*)ap3;
    __builtin_prefetch((const void*)(ap0 + 512), 0, 1);   // global_prefetch next K tile
    __builtin_prefetch((const void*)(ap2 + 512), 0, 1);
    acc0 = __builtin_amdgcn_wmma_f32_16x16x32_f16(false, a0, false, b, (short)0, acc0, false, false);
    acc1 = __builtin_amdgcn_wmma_f32_16x16x32_f16(false, a1, false, b, (short)0, acc1, false, false);
    acc2 = __builtin_amdgcn_wmma_f32_16x16x32_f16(false, a2, false, b, (short)0, acc2, false, false);
    acc3 = __builtin_amdgcn_wmma_f32_16x16x32_f16(false, a3, false, b, (short)0, acc3, false, false);
    ap0 += 512; ap1 += 512; ap2 += 512; ap3 += 512; bp += bstep;
  }

  // C/D layout: VGPR i -> row M = i + (lane<16 ? 0 : 8), col = lane&15
  const int n = tn * 16 + (lane & 15);
  const int roff = (lane < 16) ? 0 : 8;
  v8f accs[4] = {acc0, acc1, acc2, acc3};
#pragma unroll
  for (int t = 0; t < 4; ++t) {
    const int mt = mt0 + t;
    if (mt >= mtiles) break;
    float* __restrict__ crow = C + (long long)(mt * 16 + roff) * Np + n;
#pragma unroll
    for (int i = 0; i < 8; ++i) crow[(long long)i * Np] = accs[t][i];
  }
}

// ---------------- layer-1 attention logits: al[n,h] = sum_c h[n,h,c]*a[h,c] -----
__global__ void k_attn1(const float* __restrict__ h, const float* __restrict__ a_src,
                        const float* __restrict__ a_dst,
                        float* __restrict__ als, float* __restrict__ ald) {
  __shared__ float ss[C1], sd[C1];
  const int nid = blockIdx.x;
  const int t = threadIdx.x;                    // 0..255, head = t>>6
  float v = h[(long long)nid * C1 + t];
  ss[t] = v * a_src[t];
  sd[t] = v * a_dst[t];
  __syncthreads();
  for (int s = 32; s > 0; s >>= 1) {
    if ((t & 63) < s) { ss[t] += ss[t + s]; sd[t] += sd[t + s]; }
    __syncthreads();
  }
  if ((t & 63) == 0) {
    int hh = t >> 6;
    als[(long long)nid * HEADS + hh] = ss[t];
    ald[(long long)nid * HEADS + hh] = sd[t];
  }
}

// ---------------- layer-2 attention logits (H=1, 41 channels) -------------------
__global__ void k_attn2(const float* __restrict__ g, const float* __restrict__ a_src,
                        const float* __restrict__ a_dst,
                        float* __restrict__ als, float* __restrict__ ald) {
  __shared__ float ss[64], sd[64];
  const int nid = blockIdx.x;
  const int t = threadIdx.x;                    // 0..63
  float v = (t < OUT_C) ? g[(long long)nid * N2P + t] : 0.0f;
  ss[t] = (t < OUT_C) ? v * a_src[t] : 0.0f;
  sd[t] = (t < OUT_C) ? v * a_dst[t] : 0.0f;
  __syncthreads();
  for (int s = 32; s > 0; s >>= 1) {
    if (t < s) { ss[t] += ss[t + s]; sd[t] += sd[t + s]; }
    __syncthreads();
  }
  if (t == 0) { als[nid] = ss[0]; ald[nid] = sd[0]; }
}

// ---------------- ordered-uint float encoding for atomic max --------------------
__device__ __forceinline__ unsigned enc_f(float x) {
  unsigned b = __float_as_uint(x);
  return (b & 0x80000000u) ? ~b : (b | 0x80000000u);
}
__device__ __forceinline__ float dec_f(unsigned u) {
  return __uint_as_float((u & 0x80000000u) ? (u ^ 0x80000000u) : ~u);
}

template <int H>
__global__ void k_edge_logit_max(const int* __restrict__ src, const int* __restrict__ dst,
                                 const float* __restrict__ als, const float* __restrict__ ald,
                                 float* __restrict__ e, unsigned* __restrict__ mu) {
  int i = blockIdx.x * blockDim.x + threadIdx.x;
  if (i >= E_EDGES * H) return;
  int ed = i / H, hh = i % H;
  float v = als[(long long)src[ed] * H + hh] + ald[(long long)dst[ed] * H + hh];
  v = (v < 0.0f) ? v * NEG_SLOPE : v;           // leaky_relu
  e[i] = v;
  atomicMax(&mu[(long long)dst[ed] * H + hh], enc_f(v));
}

// decode max in place; mu==0 decodes to NaN (empty segment) -> 0, matching reference
__global__ void k_decode_max(unsigned* __restrict__ mu, float* __restrict__ m, int n) {
  int i = blockIdx.x * blockDim.x + threadIdx.x;
  if (i >= n) return;
  float v = dec_f(mu[i]);
  m[i] = (v * 0.0f == 0.0f) ? v : 0.0f;         // finite check without header deps
}

template <int H>
__global__ void k_edge_exp_sum(const int* __restrict__ dst, const float* __restrict__ m,
                               float* __restrict__ e, float* __restrict__ den) {
  int i = blockIdx.x * blockDim.x + threadIdx.x;
  if (i >= E_EDGES * H) return;
  int ed = i / H, hh = i % H;
  float ex = expf(e[i] - m[(long long)dst[ed] * H + hh]);
  e[i] = ex;
  atomicAdd(&den[(long long)dst[ed] * H + hh], ex);
}

__global__ void k_init_bias(float* __restrict__ out, const float* __restrict__ bias,
                            int cols, long long total) {
  long long i = (long long)blockIdx.x * blockDim.x + threadIdx.x;
  if (i >= total) return;
  out[i] = bias[i % cols];
}

// layer 1 message scatter: 256 threads handle one edge (all 4 heads x 64 channels)
__global__ void k_scatter1(const int* __restrict__ src, const int* __restrict__ dst,
                           const float* __restrict__ h, const float* __restrict__ ex,
                           const float* __restrict__ den, float* __restrict__ out) {
  const int ed = blockIdx.x;
  const int t = threadIdx.x;                    // channel in [0,256), head = t>>6
  const int s = src[ed], d = dst[ed];
  const int hh = t >> 6;
  float alpha = ex[(long long)ed * HEADS + hh] / den[(long long)d * HEADS + hh];
  atomicAdd(&out[(long long)d * C1 + t], h[(long long)s * C1 + t] * alpha);
}

// layer 2 message scatter: 64 threads per edge (41 active), 4 edges per block
__global__ void k_scatter2(const int* __restrict__ src, const int* __restrict__ dst,
                           const float* __restrict__ g, const float* __restrict__ ex,
                           const float* __restrict__ den, float* __restrict__ out) {
  long long ed = (long long)blockIdx.x * 4 + (threadIdx.x >> 6);
  int c = threadIdx.x & 63;
  if (ed >= E_EDGES || c >= OUT_C) return;
  int s = src[ed], d = dst[ed];
  float alpha = ex[ed] / den[d];
  atomicAdd(&out[(long long)d * OUT_C + c], g[(long long)s * N2P + c] * alpha);
}

// -------------------------------------------------------------------------------
extern "C" void kernel_launch(void* const* d_in, const int* in_sizes, int n_in,
                              void* d_out, int out_size, void* d_ws, size_t ws_size,
                              hipStream_t stream) {
  const float* x      = (const float*)d_in[0];
  const int*   src    = (const int*)  d_in[1];
  const int*   dst    = (const int*)  d_in[2];
  const float* W1     = (const float*)d_in[3];
  const float* a_src1 = (const float*)d_in[4];
  const float* a_dst1 = (const float*)d_in[5];
  const float* b1     = (const float*)d_in[6];
  const float* W2     = (const float*)d_in[7];
  const float* a_src2 = (const float*)d_in[8];
  const float* a_dst2 = (const float*)d_in[9];
  const float* b2     = (const float*)d_in[10];
  float* out = (float*)d_out;

  // ---- workspace carve-up (256B aligned) ----
  char* ws = (char*)d_ws;
  size_t off = 0;
  auto carve = [&](size_t bytes) -> void* {
    void* p = ws + off;
    off = (off + bytes + 255) & ~(size_t)255;
    return p;
  };
  _Float16* a1sw = (_Float16*)carve((size_t)MT * KT1 * 512 * 2);  // x, A-swizzled f16
  _Float16* b1sw = (_Float16*)carve((size_t)KT1 * NT1 * 512 * 2); // W1, B-swizzled f16
  float*    h1   = (float*)   carve((size_t)N_NODES * C1 * 4);    // x @ W1
  float*    als1 = (float*)   carve((size_t)N_NODES * HEADS * 4);
  float*    ald1 = (float*)   carve((size_t)N_NODES * HEADS * 4);
  float*    e1   = (float*)   carve((size_t)E_EDGES * HEADS * 4); // logits -> exp
  unsigned* mu1  = (unsigned*)carve((size_t)N_NODES * HEADS * 4); // enc max -> decoded float
  float*    den1 = (float*)   carve((size_t)N_NODES * HEADS * 4);
  float*    out1 = (float*)   carve((size_t)N_NODES * C1 * 4);    // layer-1 aggregated
  _Float16* a2sw = (_Float16*)carve((size_t)MT * KT2 * 512 * 2);  // elu(out1), A-swizzled
  _Float16* b2sw = (_Float16*)carve((size_t)KT2 * NT2 * 512 * 2); // W2, B-swizzled
  float*    g2   = (float*)   carve((size_t)N_NODES * N2P * 4);   // h2 @ W2 (padded cols)
  float*    als2 = (float*)   carve((size_t)N_NODES * 4);
  float*    ald2 = (float*)   carve((size_t)N_NODES * 4);
  float*    e2   = (float*)   carve((size_t)E_EDGES * 4);
  unsigned* mu2  = (unsigned*)carve((size_t)N_NODES * 4);
  float*    den2 = (float*)   carve((size_t)N_NODES * 4);

  const int TPB = 256;
  auto blocks = [](long long n, int tpb) { return (unsigned)((n + tpb - 1) / tpb); };

  // ---- swizzle + convert operands into WMMA fragment order ----
  const long long a1tot = (long long)MT * KT1 * 512;
  k_swz_a<<<blocks(a1tot, TPB), TPB, 0, stream>>>(x, a1sw, N_NODES, F_IN, KT1, a1tot);
  const long long b1tot = (long long)KT1 * NT1 * 512;
  k_swz_b<<<blocks(b1tot, TPB), TPB, 0, stream>>>(W1, b1sw, F_IN, C1, NT1, b1tot);
  const long long b2tot = (long long)KT2 * NT2 * 512;
  k_swz_b<<<blocks(b2tot, TPB), TPB, 0, stream>>>(W2, b2sw, K2P, OUT_C, NT2, b2tot);

  // ---- layer 1 GEMM: h1 = x @ W1 ----
  k_gemm_wmma_f16<<<dim3((NT1 + 7) / 8, (MT + 3) / 4), 256, 0, stream>>>(
      a1sw, b1sw, h1, MT, NT1, KT1, C1);

  // ---- layer 1 attention ----
  k_attn1<<<N_NODES, C1, 0, stream>>>(h1, a_src1, a_dst1, als1, ald1);
  hipMemsetAsync(mu1, 0, (size_t)N_NODES * HEADS * 4, stream);
  hipMemsetAsync(den1, 0, (size_t)N_NODES * HEADS * 4, stream);
  k_edge_logit_max<HEADS><<<blocks((long long)E_EDGES * HEADS, TPB), TPB, 0, stream>>>(
      src, dst, als1, ald1, e1, mu1);
  k_decode_max<<<blocks((long long)N_NODES * HEADS, TPB), TPB, 0, stream>>>(
      mu1, (float*)mu1, N_NODES * HEADS);
  k_edge_exp_sum<HEADS><<<blocks((long long)E_EDGES * HEADS, TPB), TPB, 0, stream>>>(
      dst, (const float*)mu1, e1, den1);
  k_init_bias<<<blocks((long long)N_NODES * C1, TPB), TPB, 0, stream>>>(
      out1, b1, C1, (long long)N_NODES * C1);
  k_scatter1<<<E_EDGES, C1, 0, stream>>>(src, dst, h1, e1, den1, out1);

  // ---- ELU + swizzle for layer 2 GEMM ----
  const long long a2tot = (long long)MT * KT2 * 512;
  k_elu_swz<<<blocks(a2tot, TPB), TPB, 0, stream>>>(out1, a2sw, N_NODES, C1, KT2, a2tot);

  // ---- layer 2 GEMM: g2 = h2 @ W2 ----
  k_gemm_wmma_f16<<<dim3((NT2 + 7) / 8, (MT + 3) / 4), 256, 0, stream>>>(
      a2sw, b2sw, g2, MT, NT2, KT2, N2P);

  // ---- layer 2 attention (H = 1) ----
  k_attn2<<<N_NODES, 64, 0, stream>>>(g2, a_src2, a_dst2, als2, ald2);
  hipMemsetAsync(mu2, 0, (size_t)N_NODES * 4, stream);
  hipMemsetAsync(den2, 0, (size_t)N_NODES * 4, stream);
  k_edge_logit_max<1><<<blocks((long long)E_EDGES, TPB), TPB, 0, stream>>>(
      src, dst, als2, ald2, e2, mu2);
  k_decode_max<<<blocks((long long)N_NODES, TPB), TPB, 0, stream>>>(
      mu2, (float*)mu2, N_NODES);
  k_edge_exp_sum<1><<<blocks((long long)E_EDGES, TPB), TPB, 0, stream>>>(
      dst, (const float*)mu2, e2, den2);
  k_init_bias<<<blocks((long long)N_NODES * OUT_C, TPB), TPB, 0, stream>>>(
      out, b2, OUT_C, (long long)N_NODES * OUT_C);
  k_scatter2<<<blocks((long long)E_EDGES, 4), 256, 0, stream>>>(
      src, dst, g2, e2, den2, out);
}
